// CardAwarePolicy_49675591746172
// MI455X (gfx1250) — matrix operand order repaired and checked by reading, not compiled, not run on hardware
//
#include <hip/hip_runtime.h>
#include <hip/hip_bf16.h>

typedef __attribute__((ext_vector_type(16))) _Float16 v16h;
typedef __attribute__((ext_vector_type(8)))  float    v8f;

#define DEVINL __device__ __forceinline__
// Per-wave LDS producer->consumer fence (DS ops are in-order within a wave).
#define LDS_FENCE() asm volatile("s_wait_dscnt 0" ::: "memory")

// Fast hardware reciprocal (v_rcp_f32) — ample precision for softmax/mean.
DEVINL float fast_rcp(float x) { return __builtin_amdgcn_rcpf(x); }

// ---------------- problem constants ----------------
constexpr int Bsz = 32768;

// ---------------- workspace layout (bytes) ----------------
constexpr size_t alignup(size_t x) { return (x + 255) & ~size_t(255); }
constexpr size_t O_SCORE = 0;                                        // f16 [4][54][54]
constexpr size_t O_VTAB  = alignup(O_SCORE + size_t(4*54*54)*2);     // f16 [54][64]
constexpr size_t O_AREPS = alignup(O_VTAB  + size_t(54*64)*2);       // f16 [20][64]  (arep@s1_w[:,128:].T + s1_b)
// Weights stored pre-packed in WMMA B-fragment layout: [kt][nt][lane][16 halves]
constexpr size_t O_WOUT  = alignup(O_AREPS + size_t(20*64)*2);       // f16 K=64  N=64
constexpr size_t O_WG1   = alignup(O_WOUT  + size_t(64*64)*2);       // f16 K=32  N=64 (K padded 12->32)
constexpr size_t O_WG2   = alignup(O_WG1   + size_t(32*64)*2);       // f16 K=64  N=32
constexpr size_t O_WC1   = alignup(O_WG2   + size_t(64*32)*2);       // f16 K=96  N=128
constexpr size_t O_WC2   = alignup(O_WC1   + size_t(96*128)*2);      // f16 K=128 N=128
constexpr size_t O_WS1A  = alignup(O_WC2   + size_t(128*128)*2);     // f16 K=128 N=64
constexpr size_t O_WS2   = alignup(O_WS1A  + size_t(128*64)*2);      // f16 K=64  N=32
constexpr size_t O_BIAS  = alignup(O_WS2   + size_t(64*32)*2);       // f32 [481]
// bias sub-offsets (in floats)
constexpr int BO_OUT = 0, BO_G1 = 64, BO_G2 = 128, BO_C1 = 160, BO_C2 = 288,
              BO_S2 = 416, BO_S3W = 448, BO_S3B = 480;

// Pack a logical [K][N] f16 weight into B-fragment order:
//   dst[((kt*NT + nt)*32 + lane)*16 + m] = W[kt*32 + 16*(m>=8) + 8*(lane>=16) + (m&7)][nt*16 + (lane&15)]
template <typename F>
DEVINL void pack_weight(_Float16* dst, int K, int N, int tid, F src) {
  const int NT = N >> 4;
  for (int idx = tid; idx < K * N; idx += 256) {
    int m = idx & 15;
    int lane = (idx >> 4) & 31;
    int blk = idx >> 9;
    int kt = blk / NT, nt = blk % NT;
    int k = kt * 32 + ((m >> 3) << 4) + ((lane >> 4) << 3) + (m & 7);
    int n = nt * 16 + (lane & 15);
    dst[idx] = (_Float16)src(k, n);
  }
}

// =====================================================================
// Prep kernel: qkv table -> score table, v table, arep_s, packed weights.
// =====================================================================
__global__ __launch_bounds__(256) void prep_kernel(
    const float* __restrict__ emb, const float* __restrict__ ipw,
    const float* __restrict__ ipb,
    const float* __restrict__ out_w, const float* __restrict__ out_b,
    const float* __restrict__ g1w, const float* __restrict__ g1b,
    const float* __restrict__ g2w, const float* __restrict__ g2b,
    const float* __restrict__ c1w, const float* __restrict__ c1b,
    const float* __restrict__ c2w, const float* __restrict__ c2b,
    const float* __restrict__ s1w, const float* __restrict__ s1b,
    const float* __restrict__ s2w, const float* __restrict__ s2b,
    const float* __restrict__ s3w, const float* __restrict__ s3b,
    const int* __restrict__ aci, const int* __restrict__ acc,
    char* __restrict__ ws)
{
  __shared__ float qkv[54 * 192];   // (emb @ in_proj_w.T + b)
  __shared__ float arep[20 * 64];   // masked-mean action card embedding
  const int tid = threadIdx.x;

  for (int idx = tid; idx < 54 * 192; idx += 256) {
    int c = idx / 192, f = idx % 192;
    float s = ipb[f];
    for (int e = 0; e < 64; ++e) s += emb[c * 64 + e] * ipw[f * 64 + e];
    qkv[idx] = s;
  }
  for (int idx = tid; idx < 20 * 64; idx += 256) {
    int a = idx >> 6, e = idx & 63;
    int cnt = acc[a];
    float s = 0.f;
    for (int k = 0; k < 4; ++k)
      if (k < cnt) s += emb[aci[a * 4 + k] * 64 + e];
    arep[idx] = s / (float)(cnt > 1 ? cnt : 1);   // prep is tiny; exact div
  }
  __syncthreads();

  // score_tab[h][ci][cj] = (q_h(ci) . k_h(cj)) / sqrt(16)
  _Float16* sc = (_Float16*)(ws + O_SCORE);
  for (int idx = tid; idx < 4 * 54 * 54; idx += 256) {
    int h = idx / (54 * 54), r = idx % (54 * 54), ci = r / 54, cj = r % 54;
    float s = 0.f;
    for (int d = 0; d < 16; ++d)
      s += qkv[ci * 192 + h * 16 + d] * qkv[cj * 192 + 64 + h * 16 + d];
    sc[idx] = (_Float16)(s * 0.25f);
  }
  _Float16* vt = (_Float16*)(ws + O_VTAB);
  for (int idx = tid; idx < 54 * 64; idx += 256)
    vt[idx] = (_Float16)qkv[(idx >> 6) * 192 + 128 + (idx & 63)];

  // arep_s[a][n] = arep[a] . s1_w[n, 128:192] + s1_b[n]
  _Float16* ar = (_Float16*)(ws + O_AREPS);
  for (int idx = tid; idx < 20 * 64; idx += 256) {
    int a = idx >> 6, n = idx & 63;
    float s = s1b[n];
    for (int e = 0; e < 64; ++e) s += arep[a * 64 + e] * s1w[n * 192 + 128 + e];
    ar[idx] = (_Float16)s;
  }

  // Weights pre-packed into B-fragment layout (source is torch [N][K]).
  pack_weight((_Float16*)(ws + O_WOUT), 64, 64, tid,
              [=](int k, int n) { return out_w[n * 64 + k]; });
  pack_weight((_Float16*)(ws + O_WG1), 32, 64, tid,
              [=](int k, int n) { return k < 12 ? g1w[n * 12 + k] : 0.f; });
  pack_weight((_Float16*)(ws + O_WG2), 64, 32, tid,
              [=](int k, int n) { return g2w[n * 64 + k]; });
  pack_weight((_Float16*)(ws + O_WC1), 96, 128, tid,
              [=](int k, int n) { return c1w[n * 96 + k]; });
  pack_weight((_Float16*)(ws + O_WC2), 128, 128, tid,
              [=](int k, int n) { return c2w[n * 128 + k]; });
  pack_weight((_Float16*)(ws + O_WS1A), 128, 64, tid,
              [=](int k, int n) { return s1w[n * 192 + k]; });
  pack_weight((_Float16*)(ws + O_WS2), 64, 32, tid,
              [=](int k, int n) { return s2w[n * 64 + k]; });

  float* bb = (float*)(ws + O_BIAS);
  for (int i = tid; i < 64; i += 256)  bb[BO_OUT + i] = 8.f * out_b[i];  // sum over S=8
  for (int i = tid; i < 64; i += 256)  bb[BO_G1 + i] = g1b[i];
  for (int i = tid; i < 32; i += 256)  bb[BO_G2 + i] = g2b[i];
  for (int i = tid; i < 128; i += 256) bb[BO_C1 + i] = c1b[i];
  for (int i = tid; i < 128; i += 256) bb[BO_C2 + i] = c2b[i];
  for (int i = tid; i < 32; i += 256)  bb[BO_S2 + i] = s2b[i];
  for (int i = tid; i < 32; i += 256)  bb[BO_S3W + i] = s3w[i];
  if (tid == 0) bb[BO_S3B] = s3b[0];
}

// =====================================================================
// WMMA fragment helpers (CDNA5 ISA 7.12.2 layouts, wave32)
// =====================================================================
// A 16x32 f16 from row-major LDS tile: lanes 0-15 -> M=lane, K={0..7,16..23};
// lanes 16-31 same rows, K shifted by 8. Two contiguous 16B runs -> ds_load_b128 x2.
DEVINL v16h load_A_frag(const _Float16* t, int ld, int kbase, int lane) {
  const int row = lane & 15;
  const int off = (lane >> 4) << 3;
  const _Float16* p = t + row * ld + kbase;
  v16h a;
#pragma unroll
  for (int m = 0; m < 8; ++m) a[m] = p[off + m];
#pragma unroll
  for (int m = 0; m < 8; ++m) a[8 + m] = p[16 + off + m];
  return a;
}
// Pre-packed B fragment: one aligned 32-byte load per lane (2x global_load_b128).
DEVINL v16h load_B_packed(const _Float16* base, int blk, int lane) {
  return ((const v16h*)base)[blk * 32 + lane];
}

// D = [relu]([scale_row] (A*B [+ bias])), stored as f16 to an LDS tile.
// All policy knobs are compile-time -> branch-free epilogue.
template <int KS, int NT, bool BIAS, bool RELU, bool SCALE>
DEVINL void wmma_nt(const v16h* A, const _Float16* Wp, const float* bias,
                    const float* rs /*[8] per-row scales in regs*/,
                    _Float16* outT, int outld, int lane) {
  const int col = lane & 15;
  const int rbase = (lane >> 4) << 3;
#pragma unroll
  for (int nt = 0; nt < NT; ++nt) {
    v8f acc;
    const float bv = BIAS ? bias[nt * 16 + col] : 0.f;
#pragma unroll
    for (int r = 0; r < 8; ++r) acc[r] = bv;
#pragma unroll
    for (int ks = 0; ks < KS; ++ks) {
      v16h b = load_B_packed(Wp, ks * NT + nt, lane);
      acc = __builtin_amdgcn_wmma_f32_16x16x32_f16(false, A[ks], false, b,
                                                   (short)0, acc, false, false);
    }
#pragma unroll
    for (int r = 0; r < 8; ++r) {
      float v = acc[r];
      if (RELU) v = fmaxf(v, 0.f);
      if (SCALE) v *= rs[r];
      outT[(rbase + r) * outld + nt * 16 + col] = (_Float16)v;
    }
  }
}

// =====================================================================
// Main fused kernel: 128 threads = 4 wave32, 16 batch rows per wave.
// =====================================================================
__global__ __launch_bounds__(128) void policy_main(
    const int* __restrict__ hand_cards, const float* __restrict__ game_state,
    const int* __restrict__ hand_size, const int* __restrict__ nva_p,
    const char* __restrict__ ws, float* __restrict__ out)
{
  __shared__ _Float16 sVtab[54 * 64];
  __shared__ _Float16 sArep[20 * 64];
  __shared__ _Float16 sTileA[4][16 * 128];
  __shared__ _Float16 sTileB[4][16 * 128];
  __shared__ float    sScores[4][16 * 20];

  const int tid = threadIdx.x;
  // Warm score table into cache (L2-resident, 23KB, reused by every row).
  __builtin_prefetch(ws + O_SCORE + (size_t)tid * 128, 0, 1);

  const _Float16* gv = (const _Float16*)(ws + O_VTAB);
  for (int i = tid; i < 54 * 64; i += 128) sVtab[i] = gv[i];
  const _Float16* ga = (const _Float16*)(ws + O_AREPS);
  for (int i = tid; i < 20 * 64; i += 128) sArep[i] = ga[i];
  __syncthreads();

  const int wid = tid >> 5, lane = tid & 31;
  const int r0 = (blockIdx.x * 4 + wid) * 16;
  _Float16* tA = sTileA[wid];
  _Float16* tB = sTileB[wid];
  float* scoreW = sScores[wid];

  const int col = lane & 15;
  const int rbase = (lane >> 4) << 3;

  // Per-row 1/hand_len directly into registers (8 rows owned by this lane).
  float rs[8];
#pragma unroll
  for (int r = 0; r < 8; ++r) {
    int hs = hand_size[r0 + rbase + r];
    rs[r] = fast_rcp((float)(hs > 1 ? hs : 1));
  }

  // game_state -> tA[:,96:128] (K padded 12->32 with zeros)
  for (int idx = lane; idx < 16 * 32; idx += 32) {
    int r = idx >> 5, c = idx & 31;
    tA[r * 128 + 96 + c] = (_Float16)(c < 12 ? game_state[(r0 + r) * 12 + c] : 0.f);
  }

  // ---------- attention via tables; pooled over queries -> tA[:,0:64] ----------
  {
    const int row = lane & 15, half = lane >> 4;   // 2 lanes/row, 2 heads/lane
    const int b = r0 + row;
    int cards[8];
    const int4* hc4 = (const int4*)(hand_cards + b * 8);
    int4 lo = hc4[0], hi = hc4[1];
    cards[0] = lo.x; cards[1] = lo.y; cards[2] = lo.z; cards[3] = lo.w;
    cards[4] = hi.x; cards[5] = hi.y; cards[6] = hi.z; cards[7] = hi.w;

    const _Float16* st = (const _Float16*)(ws + O_SCORE);
    float w0[8], w1[8];
#pragma unroll
    for (int j = 0; j < 8; ++j) { w0[j] = 0.f; w1[j] = 0.f; }

#pragma unroll
    for (int hh = 0; hh < 2; ++hh) {
      const int h = half * 2 + hh;
      const _Float16* sth = st + h * 54 * 54;
      float* wacc = hh ? w1 : w0;
      for (int i = 0; i < 8; ++i) {                 // all S query positions (incl. pads)
        const _Float16* sr = sth + cards[i] * 54;
        float s[8], mx = -3.4e38f;
#pragma unroll
        for (int j = 0; j < 8; ++j) {
          float v = (cards[j] != 0) ? (float)sr[cards[j]] : -1e9f;  // key padding mask
          s[j] = v; mx = fmaxf(mx, v);
        }
        float sum = 0.f;
#pragma unroll
        for (int j = 0; j < 8; ++j) { s[j] = __expf(s[j] - mx); sum += s[j]; }
        const float inv = fast_rcp(sum);            // v_rcp_f32, not IEEE div
#pragma unroll
        for (int j = 0; j < 8; ++j) wacc[j] += s[j] * inv;          // sum_i attn[h,i,j]
      }
    }
    float p[32];
#pragma unroll
    for (int d = 0; d < 32; ++d) p[d] = 0.f;
#pragma unroll
    for (int j = 0; j < 8; ++j) {
      const _Float16* vr = sVtab + cards[j] * 64 + half * 32;
#pragma unroll
      for (int d = 0; d < 32; ++d)
        p[d] += ((d < 16) ? w0[j] : w1[j]) * (float)vr[d];
    }
#pragma unroll
    for (int d = 0; d < 32; ++d) tA[row * 128 + half * 32 + d] = (_Float16)p[d];
  }
  LDS_FENCE();

  const float* bias = (const float*)(ws + O_BIAS);

  // 1) hand_ctx = (pooled @ out_w.T + 8*out_b) / hand_len  -> tB[:,0:64]
  { v16h A[2] = { load_A_frag(tA, 128, 0, lane), load_A_frag(tA, 128, 32, lane) };
    wmma_nt<2, 4, true, false, true>(A, (const _Float16*)(ws + O_WOUT),
                                     bias + BO_OUT, rs, tB, 128, lane); }
  LDS_FENCE();
  // 2) g1: relu(gs @ g1_w.T + b)  -> tB[:,64:128]
  { v16h A[1] = { load_A_frag(tA, 128, 96, lane) };
    wmma_nt<1, 4, true, true, false>(A, (const _Float16*)(ws + O_WG1),
                                     bias + BO_G1, rs, tB + 64, 128, lane); }
  LDS_FENCE();
  // 3) g2 -> tA[:,0:32]
  { v16h A[2] = { load_A_frag(tB, 128, 64, lane), load_A_frag(tB, 128, 96, lane) };
    wmma_nt<2, 2, true, true, false>(A, (const _Float16*)(ws + O_WG2),
                                     bias + BO_G2, rs, tA, 128, lane); }
  LDS_FENCE();
  // 4) c1 over concat[hand_ctx(tB 0:64) | g(tA 0:32)] -> tA[:,0:128]
  { v16h A[3] = { load_A_frag(tB, 128, 0, lane), load_A_frag(tB, 128, 32, lane),
                  load_A_frag(tA, 128, 0, lane) };
    wmma_nt<3, 8, true, true, false>(A, (const _Float16*)(ws + O_WC1),
                                     bias + BO_C1, rs, tA, 128, lane); }
  LDS_FENCE();
  // 5) c2 -> tB[:,0:128]
  { v16h A[4] = { load_A_frag(tA, 128, 0, lane), load_A_frag(tA, 128, 32, lane),
                  load_A_frag(tA, 128, 64, lane), load_A_frag(tA, 128, 96, lane) };
    wmma_nt<4, 8, true, true, false>(A, (const _Float16*)(ws + O_WC2),
                                     bias + BO_C2, rs, tB, 128, lane); }
  LDS_FENCE();
  // 6) ctx_s = ctx @ s1_w[:, :128].T (bias folded into arep_s) -> tA[:,0:64]
  { v16h A[4] = { load_A_frag(tB, 128, 0, lane), load_A_frag(tB, 128, 32, lane),
                  load_A_frag(tB, 128, 64, lane), load_A_frag(tB, 128, 96, lane) };
    wmma_nt<4, 4, false, false, false>(A, (const _Float16*)(ws + O_WS1A),
                                       nullptr, rs, tA, 128, lane); }
  LDS_FENCE();

  // 7) scorer: per action, h = relu(ctx_s + arep_s[a]) built directly in
  //    A-fragment layout, four 16x16x32 WMMAs, VALU s3 dot + shfl reduce.
  {
    const v16h ctxA0 = load_A_frag(tA, 128, 0, lane);
    const v16h ctxA1 = load_A_frag(tA, 128, 32, lane);
    const _Float16* w2 = (const _Float16*)(ws + O_WS2);   // packed, NT=2
    const v16h B00 = load_B_packed(w2, 0, lane);  // ks=0, nt=0
    const v16h B01 = load_B_packed(w2, 1, lane);  // ks=0, nt=1
    const v16h B10 = load_B_packed(w2, 2, lane);  // ks=1, nt=0
    const v16h B11 = load_B_packed(w2, 3, lane);  // ks=1, nt=1

    const int off = rbase;              // A-frag K offset == D-frag row base
    const float s3v0 = bias[BO_S3W + col];
    const float s3v1 = bias[BO_S3W + 16 + col];
    const float s3bv = bias[BO_S3B];
    const float s2b0 = bias[BO_S2 + col];
    const float s2b1 = bias[BO_S2 + 16 + col];
    const int nva = *nva_p;

    for (int a = 0; a < 20; ++a) {
      const _Float16* arw = sArep + a * 64;
      v16h h0, h1;
#pragma unroll
      for (int m = 0; m < 8; ++m) {
        h0[m]     = (_Float16)fmaxf((float)ctxA0[m]     + (float)arw[off + m], 0.f);
        h0[8 + m] = (_Float16)fmaxf((float)ctxA0[8 + m] + (float)arw[16 + off + m], 0.f);
        h1[m]     = (_Float16)fmaxf((float)ctxA1[m]     + (float)arw[32 + off + m], 0.f);
        h1[8 + m] = (_Float16)fmaxf((float)ctxA1[8 + m] + (float)arw[48 + off + m], 0.f);
      }
      v8f acc0, acc1;
#pragma unroll
      for (int r = 0; r < 8; ++r) { acc0[r] = s2b0; acc1[r] = s2b1; }
      acc0 = __builtin_amdgcn_wmma_f32_16x16x32_f16(false, h0, false, B00, (short)0, acc0, false, false);
      acc0 = __builtin_amdgcn_wmma_f32_16x16x32_f16(false, h1, false, B10, (short)0, acc0, false, false);
      acc1 = __builtin_amdgcn_wmma_f32_16x16x32_f16(false, h0, false, B01, (short)0, acc1, false, false);
      acc1 = __builtin_amdgcn_wmma_f32_16x16x32_f16(false, h1, false, B11, (short)0, acc1, false, false);
#pragma unroll
      for (int r = 0; r < 8; ++r) {
        float v = fmaxf(acc0[r], 0.f) * s3v0 + fmaxf(acc1[r], 0.f) * s3v1;
        v += __shfl_xor(v, 1, 16);
        v += __shfl_xor(v, 2, 16);
        v += __shfl_xor(v, 4, 16);
        v += __shfl_xor(v, 8, 16);
        v += s3bv;
        if (a >= nva) v = -1e8f;
        if (col == 0) scoreW[(off + r) * 20 + a] = v;
      }
    }
  }
  LDS_FENCE();
  for (int idx = lane; idx < 16 * 20; idx += 32)
    out[(r0 + idx / 20) * 20 + idx % 20] = scoreW[idx];
}

// =====================================================================
extern "C" void kernel_launch(void* const* d_in, const int* in_sizes, int n_in,
                              void* d_out, int out_size, void* d_ws, size_t ws_size,
                              hipStream_t stream) {
  const int*   hand_cards = (const int*)d_in[0];
  const float* game_state = (const float*)d_in[1];
  const int*   hand_size  = (const int*)d_in[2];
  const int*   aci        = (const int*)d_in[3];
  const int*   acc        = (const int*)d_in[4];
  const int*   nva        = (const int*)d_in[5];
  const float* emb        = (const float*)d_in[6];
  const float* ipw        = (const float*)d_in[7];
  const float* ipb        = (const float*)d_in[8];
  const float* out_w      = (const float*)d_in[9];
  const float* out_b      = (const float*)d_in[10];
  const float* g1w        = (const float*)d_in[11];
  const float* g1b        = (const float*)d_in[12];
  const float* g2w        = (const float*)d_in[13];
  const float* g2b        = (const float*)d_in[14];
  const float* c1w        = (const float*)d_in[15];
  const float* c1b        = (const float*)d_in[16];
  const float* c2w        = (const float*)d_in[17];
  const float* c2b        = (const float*)d_in[18];
  const float* s1w        = (const float*)d_in[19];
  const float* s1b        = (const float*)d_in[20];
  const float* s2w        = (const float*)d_in[21];
  const float* s2b        = (const float*)d_in[22];
  const float* s3w        = (const float*)d_in[23];
  const float* s3b        = (const float*)d_in[24];

  char* ws = (char*)d_ws;
  prep_kernel<<<1, 256, 0, stream>>>(emb, ipw, ipb, out_w, out_b, g1w, g1b,
                                     g2w, g2b, c1w, c1b, c2w, c2b, s1w, s1b,
                                     s2w, s2b, s3w, s3b, aci, acc, ws);

  const int rows_per_block = 4 * 16;  // 4 wave32 x 16 rows
  policy_main<<<Bsz / rows_per_block, 128, 0, stream>>>(
      hand_cards, game_state, hand_size, nva, (const char*)ws, (float*)d_out);
}
